// SparseMOE_5068061409421
// MI455X (gfx1250) — compile-verified, matrix operand order checked by reading, not run on hardware
//
#include <hip/hip_runtime.h>
#include <hip/hip_bf16.h>

// Problem constants (match reference)
constexpr int Bb = 4, Ss = 2048, Dd = 1024, Ff = 4096, Ee = 8;
constexpr int NT = Bb * Ss;           // 8192 tokens
constexpr int MT = 128;               // token tile (M): 4 waves x 32 rows
constexpr int NTL = 64;               // output-feature tile (N): 2 wave cols x 32
constexpr int KT = 32;                // bf16 WMMA K per step
constexpr int ASTR = 40;              // LDS row stride in bf16 elems (80B, 16B aligned, bank-spread)

typedef __attribute__((ext_vector_type(16))) __bf16 v16bf;
typedef __attribute__((ext_vector_type(8)))  float  v8f;
typedef __attribute__((ext_vector_type(4)))  int    v4i;

#if __has_builtin(__builtin_amdgcn_global_load_async_to_lds_b128)
#define HAS_ASYNC_LDS 1
typedef __attribute__((address_space(1))) v4i* gv4i_p;   // global (AS1) int4*
typedef __attribute__((address_space(3))) v4i* lv4i_p;   // LDS (AS3) int4*
#else
#define HAS_ASYNC_LDS 0
#endif

__device__ inline void wait_async0() {
#if __has_builtin(__builtin_amdgcn_s_wait_asynccnt)
    __builtin_amdgcn_s_wait_asynccnt(0);
#else
    asm volatile("s_wait_asynccnt 0" ::: "memory");
#endif
}

// ---------- helpers ----------

// Load a 16x32 bf16 WMMA A/B fragment from an LDS tile (row-major, stride ASTR).
// Lane L<16 : row=L,    K = 0..7 and 16..23
// Lane L>=16: row=L-16, K = 8..15 and 24..31
__device__ inline v16bf ld_frag(const __bf16* base, int row, int lane) {
    int kh = (lane & 16) ? 8 : 0;
    const __bf16* p = base + row * ASTR + kh;
    union { uint4 u[2]; v16bf v; } r;
    r.u[0] = *(const uint4*)(p);
    r.u[1] = *(const uint4*)(p + 16);
    return r.v;
}

// Convert 8 f32 -> 8 bf16 and store as one 16B LDS write.
__device__ inline void cvt8(__bf16* dst, float4 a, float4 b) {
    union { __bf16 h[8]; uint4 u; } t;
    t.h[0] = (__bf16)a.x; t.h[1] = (__bf16)a.y; t.h[2] = (__bf16)a.z; t.h[3] = (__bf16)a.w;
    t.h[4] = (__bf16)b.x; t.h[5] = (__bf16)b.y; t.h[6] = (__bf16)b.z; t.h[7] = (__bf16)b.w;
    *(uint4*)dst = t.u;
}

__device__ inline v8f wmma_bf16(v16bf a, v16bf b, v8f c) {
    return __builtin_amdgcn_wmma_f32_16x16x32_bf16(false, a, false, b, (short)0, c, false, false);
}

// ---------- kernel 0: zero per-expert counters ----------
__global__ void moe_zero_cnt(int* __restrict__ cnt) {
    if (threadIdx.x < Ee) cnt[threadIdx.x] = 0;
}

// ---------- kernel 1: router (softmax + top-2 + scatter) ----------
__global__ __launch_bounds__(256)
void moe_router(const float* __restrict__ x, const float* __restrict__ rw,
                int* __restrict__ cnt, int* __restrict__ tok, float* __restrict__ gate) {
    __shared__ float s_rw[Ee * Dd];                     // 32 KB
    int tid = threadIdx.x;
    for (int i = tid * 4; i < Ee * Dd; i += 256 * 4)
        *(float4*)&s_rw[i] = *(const float4*)&rw[i];
    __syncthreads();

    int t = blockIdx.x * 256 + tid;
    const float* xr = x + (size_t)t * Dd;
    float acc[Ee];
#pragma unroll
    for (int e = 0; e < Ee; ++e) acc[e] = 0.f;
    for (int d = 0; d < Dd; d += 4) {
        float4 xv = *(const float4*)&xr[d];
#pragma unroll
        for (int e = 0; e < Ee; ++e) {
            const float* w = &s_rw[e * Dd + d];
            acc[e] += xv.x * w[0] + xv.y * w[1] + xv.z * w[2] + xv.w * w[3];
        }
    }
    float m = acc[0];
#pragma unroll
    for (int e = 1; e < Ee; ++e) m = fmaxf(m, acc[e]);
    float p[Ee], s = 0.f;
#pragma unroll
    for (int e = 0; e < Ee; ++e) { p[e] = __expf(acc[e] - m); s += p[e]; }
    float inv = 1.0f / s;
    int e0 = 0;
#pragma unroll
    for (int e = 1; e < Ee; ++e) if (p[e] > p[e0]) e0 = e;      // first-index ties, like jax
    int e1 = (e0 == 0) ? 1 : 0;
#pragma unroll
    for (int e = 0; e < Ee; ++e) if (e != e0 && p[e] > p[e1]) e1 = e;

    int   es[2] = { e0, e1 };
    float gs[2] = { p[e0] * inv, p[e1] * inv };
#pragma unroll
    for (int k = 0; k < 2; ++k) {
        int e = es[k];
        int pos = atomicAdd(&cnt[e], 1);
        tok[e * NT + pos]  = t | (k << 30);                     // token + slot id
        gate[e * NT + pos] = gs[k];
    }
}

// ---------- kernel 2: H = gelu(X Wg^T + bg) * (X W1^T + b1), per expert, gathered rows ----------
// 128x64 tile, 8 waves each computing 32(M)x32(N) per B-matrix; register-pipelined staging,
// double-buffered LDS, one barrier per k-step.
__global__ __launch_bounds__(256)
void moe_ffn_h(const float* __restrict__ x,
               const float* __restrict__ wg_w, const float* __restrict__ wg_b,
               const float* __restrict__ w1_w, const float* __restrict__ w1_b,
               const int* __restrict__ cnt, const int* __restrict__ tok,
               __bf16* __restrict__ H) {
    int e = blockIdx.z;
    int mtile = blockIdx.y * MT;
    int n = cnt[e];
    if (mtile >= n) return;
    int ftile = blockIdx.x * NTL;

    __shared__ __bf16 sA[2][MT * ASTR];    // 20 KB
    __shared__ __bf16 sG[2][NTL * ASTR];   // 10 KB
    __shared__ __bf16 s1[2][NTL * ASTR];   // 10 KB

    int tid = threadIdx.x;
    int lane = tid & 31, wave = tid >> 5;
    int wi = wave & 3, wj = wave >> 2;            // wave -> rows [32*wi,32*wi+32), cols [32*wj,+32)
    int arowi = tid >> 1, asег_unused = 0; (void)asег_unused;
    int aseg = (tid & 1) * 16;                    // A loader: 2 threads/row, 16 floats each
    int brow = tid >> 2, bseg = (tid & 3) * 8;    // B loader: 4 threads/row, 8 floats each

    // Clamp gathered row index: rows >= n compute garbage the epilogue never stores
    // (WMMA row m of D depends only on row m of A), so loads stay branch-free.
    int aidx = mtile + arowi; if (aidx > n - 1) aidx = n - 1;
    int ment = tok[e * NT + aidx];
    const float* arow = x + (size_t)(ment & 0x3FFFFFFF) * Dd;
    const float* grow = wg_w + ((size_t)e * Ff + ftile + brow) * Dd;
    const float* orow = w1_w + ((size_t)e * Ff + ftile + brow) * Dd;

    v8f accg[2][2] = { { {}, {} }, { {}, {} } };  // [m-subtile][n-subtile], gate proj
    v8f acc1[2][2] = { { {}, {} }, { {}, {} } };  // up proj

    float4 ra0, ra1, ra2, ra3, rg0, rg1, ro0, ro1;
    auto fetch = [&](int kt) {
        ra0 = *(const float4*)&arow[kt + aseg];
        ra1 = *(const float4*)&arow[kt + aseg + 4];
        ra2 = *(const float4*)&arow[kt + aseg + 8];
        ra3 = *(const float4*)&arow[kt + aseg + 12];
        rg0 = *(const float4*)&grow[kt + bseg];
        rg1 = *(const float4*)&grow[kt + bseg + 4];
        ro0 = *(const float4*)&orow[kt + bseg];
        ro1 = *(const float4*)&orow[kt + bseg + 4];
    };
    fetch(0);

    for (int kt = 0; kt < Dd; kt += KT) {
        int cur = (kt >> 5) & 1;
        cvt8(&sA[cur][arowi * ASTR + aseg],     ra0, ra1);
        cvt8(&sA[cur][arowi * ASTR + aseg + 8], ra2, ra3);
        cvt8(&sG[cur][brow * ASTR + bseg], rg0, rg1);
        cvt8(&s1[cur][brow * ASTR + bseg], ro0, ro1);
        __syncthreads();                                   // publish buffer `cur`
        if (kt + KT < Dd) fetch(kt + KT);                  // overlap next loads with WMMA
        if (kt + 2 * KT < Dd) {
            __builtin_prefetch(&grow[kt + 2 * KT + bseg], 0, 0);
            __builtin_prefetch(&orow[kt + 2 * KT + bseg], 0, 0);
        }
        v16bf gF0 = ld_frag(&sG[cur][0], wj * 32 +      (lane & 15), lane);
        v16bf gF1 = ld_frag(&sG[cur][0], wj * 32 + 16 + (lane & 15), lane);
        v16bf oF0 = ld_frag(&s1[cur][0], wj * 32 +      (lane & 15), lane);
        v16bf oF1 = ld_frag(&s1[cur][0], wj * 32 + 16 + (lane & 15), lane);
#pragma unroll
        for (int mi = 0; mi < 2; ++mi) {
            v16bf aF = ld_frag(&sA[cur][0], wi * 32 + mi * 16 + (lane & 15), lane);
            accg[mi][0] = wmma_bf16(aF, gF0, accg[mi][0]);
            accg[mi][1] = wmma_bf16(aF, gF1, accg[mi][1]);
            acc1[mi][0] = wmma_bf16(aF, oF0, acc1[mi][0]);
            acc1[mi][1] = wmma_bf16(aF, oF1, acc1[mi][1]);
        }
    }

    int nlo = lane & 15;
    int mhalf = (lane & 16) ? 8 : 0;
#pragma unroll
    for (int mi = 0; mi < 2; ++mi) {
#pragma unroll
        for (int j = 0; j < 2; ++j) {
            int f = ftile + wj * 32 + j * 16 + nlo;
            float bg = wg_b[e * Ff + f];
            float b1 = w1_b[e * Ff + f];
            v8f cg = accg[mi][j];
            v8f c1 = acc1[mi][j];
#pragma unroll
            for (int r = 0; r < 8; ++r) {
                int mrow = wi * 32 + mi * 16 + mhalf + r;
                if (mtile + mrow < n) {
                    int ent   = tok[e * NT + mtile + mrow];
                    int token = ent & 0x3FFFFFFF;
                    int slot  = (ent >> 30) & 1;
                    float gsum = cg[r] + bg;
                    float gl = 0.5f * gsum * (1.0f + erff(gsum * 0.70710678118654752f)); // exact gelu
                    float hv = gl * (c1[r] + b1);
                    H[((size_t)slot * NT + token) * Ff + f] = (__bf16)hv;
                }
            }
        }
    }
}

// ---------- kernel 3: partial[slot,token,:] = gate * (H W2^T + b2) ----------
// A (H, already bf16) staged with async global->LDS; B register-pipelined. 128x64 tile.
__global__ __launch_bounds__(256)
void moe_ffn_out(const __bf16* __restrict__ H,
                 const float* __restrict__ w2_w, const float* __restrict__ w2_b,
                 const int* __restrict__ cnt, const int* __restrict__ tok,
                 const float* __restrict__ gate,
                 float* __restrict__ part) {
    int e = blockIdx.z;
    int mtile = blockIdx.y * MT;
    int n = cnt[e];
    if (mtile >= n) return;
    int dtile = blockIdx.x * NTL;

    __shared__ __bf16 sA[2][MT * ASTR];    // 20 KB
    __shared__ __bf16 sB[2][NTL * ASTR];   // 10 KB

    int tid = threadIdx.x;
    int lane = tid & 31, wave = tid >> 5;
    int wi = wave & 3, wj = wave >> 2;
    int arowi = tid >> 1, aseg = (tid & 1) * 16;  // A loader: 2 threads/row, 16 bf16 (2x b128)
    int browi = tid >> 2, bseg = (tid & 3) * 8;   // B loader: 4 threads/row, 8 floats

    int aidx = mtile + arowi; if (aidx > n - 1) aidx = n - 1;   // clamp, see pass 1
    int ment = tok[e * NT + aidx];
    const __bf16* arow = H + ((size_t)((ment >> 30) & 1) * NT + (ment & 0x3FFFFFFF)) * Ff;
    const float*  brow = w2_w + ((size_t)e * Dd + dtile + browi) * Ff;

    v8f acc[2][2] = { { {}, {} }, { {}, {} } };

    float4 rb0, rb1;
    auto fetchB = [&](int kt) {
        rb0 = *(const float4*)&brow[kt + bseg];
        rb1 = *(const float4*)&brow[kt + bseg + 4];
    };
    // Gathered bf16 A rows go straight into LDS (ASYNCcnt path on CDNA5).
    auto issueA = [&](int kt, int buf) {
#if HAS_ASYNC_LDS
        __builtin_amdgcn_global_load_async_to_lds_b128(
            (gv4i_p)(arow + kt + aseg),
            (lv4i_p)&sA[buf][arowi * ASTR + aseg], 0, 0);
        __builtin_amdgcn_global_load_async_to_lds_b128(
            (gv4i_p)(arow + kt + aseg + 8),
            (lv4i_p)&sA[buf][arowi * ASTR + aseg + 8], 0, 0);
#else
        *(uint4*)&sA[buf][arowi * ASTR + aseg]     = *(const uint4*)(arow + kt + aseg);
        *(uint4*)&sA[buf][arowi * ASTR + aseg + 8] = *(const uint4*)(arow + kt + aseg + 8);
#endif
    };

    fetchB(0);
    issueA(0, 0);

    for (int kt = 0; kt < Ff; kt += KT) {
        int cur = (kt >> 5) & 1;
        cvt8(&sB[cur][browi * ASTR + bseg], rb0, rb1);
#if HAS_ASYNC_LDS
        wait_async0();                                    // A async for `cur` complete
#endif
        __syncthreads();                                  // publish buffer `cur`
        if (kt + KT < Ff) {
            fetchB(kt + KT);                              // overlap with WMMA below
            issueA(kt + KT, cur ^ 1);                     // safe: `cur^1` readers retired
        }
        if (kt + 2 * KT < Ff)
            __builtin_prefetch(&brow[kt + 2 * KT + bseg], 0, 0);
        v16bf bF0 = ld_frag(&sB[cur][0], wj * 32 +      (lane & 15), lane);
        v16bf bF1 = ld_frag(&sB[cur][0], wj * 32 + 16 + (lane & 15), lane);
#pragma unroll
        for (int mi = 0; mi < 2; ++mi) {
            v16bf aF = ld_frag(&sA[cur][0], wi * 32 + mi * 16 + (lane & 15), lane);
            acc[mi][0] = wmma_bf16(aF, bF0, acc[mi][0]);
            acc[mi][1] = wmma_bf16(aF, bF1, acc[mi][1]);
        }
    }

    int nlo = lane & 15;
    int mhalf = (lane & 16) ? 8 : 0;
#pragma unroll
    for (int mi = 0; mi < 2; ++mi) {
#pragma unroll
        for (int j = 0; j < 2; ++j) {
            int d = dtile + wj * 32 + j * 16 + nlo;
            float b2 = w2_b[e * Dd + d];
            v8f c = acc[mi][j];
#pragma unroll
            for (int r = 0; r < 8; ++r) {
                int mrow = wi * 32 + mi * 16 + mhalf + r;
                if (mtile + mrow < n) {
                    int ent   = tok[e * NT + mtile + mrow];
                    int token = ent & 0x3FFFFFFF;
                    int slot  = (ent >> 30) & 1;
                    float g   = gate[e * NT + mtile + mrow];
                    part[((size_t)slot * NT + token) * Dd + d] = (c[r] + b2) * g;
                }
            }
        }
    }
}

// ---------- kernel 4: out = partial_slot0 + partial_slot1 ----------
__global__ __launch_bounds__(256)
void moe_sum(const float* __restrict__ part, float* __restrict__ out) {
    size_t i = ((size_t)blockIdx.x * 256 + threadIdx.x) * 4;
    float4 a = *(const float4*)&part[i];
    float4 b = *(const float4*)&part[(size_t)NT * Dd + i];
    float4 r; r.x = a.x + b.x; r.y = a.y + b.y; r.z = a.z + b.z; r.w = a.w + b.w;
    *(float4*)&out[i] = r;
}

// ---------- host entry ----------
extern "C" void kernel_launch(void* const* d_in, const int* in_sizes, int n_in,
                              void* d_out, int out_size, void* d_ws, size_t ws_size,
                              hipStream_t stream) {
    const float* x        = (const float*)d_in[0];
    const float* router_w = (const float*)d_in[1];
    const float* w1_w     = (const float*)d_in[2];
    const float* w1_b     = (const float*)d_in[3];
    const float* w2_w     = (const float*)d_in[4];
    const float* w2_b     = (const float*)d_in[5];
    const float* wg_w     = (const float*)d_in[6];
    const float* wg_b     = (const float*)d_in[7];
    float* out = (float*)d_out;

    // workspace layout (bytes):
    //   [0, 32)                      per-expert counters
    //   [1024, +E*NT*4)              token|slot lists
    //   [.., +E*NT*4)                gate lists
    //   [.., +2*NT*Dd*4)             per-slot output partials (64 MB)
    //   [.., +2*NT*Ff*2)             H activations, bf16 (134 MB)
    char* ws = (char*)d_ws;
    size_t o_tok  = 1024;
    size_t o_gate = o_tok  + (size_t)Ee * NT * 4;
    size_t o_part = o_gate + (size_t)Ee * NT * 4;
    size_t o_H    = o_part + (size_t)2 * NT * Dd * 4;
    int*    cnt  = (int*)ws;
    int*    tok  = (int*)(ws + o_tok);
    float*  gate = (float*)(ws + o_gate);
    float*  part = (float*)(ws + o_part);
    __bf16* H    = (__bf16*)(ws + o_H);

    moe_zero_cnt<<<1, 32, 0, stream>>>(cnt);
    moe_router<<<NT / 256, 256, 0, stream>>>(x, router_w, cnt, tok, gate);
    moe_ffn_h<<<dim3(Ff / NTL, NT / MT, Ee), 256, 0, stream>>>(x, wg_w, wg_b, w1_w, w1_b, cnt, tok, H);
    moe_ffn_out<<<dim3(Dd / NTL, NT / MT, Ee), 256, 0, stream>>>(H, w2_w, w2_b, cnt, tok, gate, part);
    moe_sum<<<(NT * Dd / 4) / 256, 256, 0, stream>>>(part, out);
}
